// Complex_Valued_Fully_Connected_Layer_33792802685864
// MI455X (gfx1250) — compile-verified, matrix operand order checked by reading, not run on hardware
//
#include <hip/hip_runtime.h>

// ---------------------------------------------------------------------------
// Complex FC layer via 3-multiplication (Karatsuba) complex GEMM:
//   t1 = x_re W_re^T, t2 = x_im W_im^T, t3 = (x_re+x_im)(W_re+W_im)^T
//   out_re = t1 - t2 ; out_im = t3 - t1 - t2
// 25% fewer v_wmma_f32_16x16x32_bf16 ops than the 4-GEMM form (XDL-bound).
// fp32 -> bf16 conversion + fp32 sum during LDS staging (single rounding).
// Wave32; 16 waves / 512 threads per block; 32x32 output per wave.
// ---------------------------------------------------------------------------

typedef __bf16 bf16_t;
typedef __attribute__((ext_vector_type(16))) __bf16 bf16x16;
typedef __attribute__((ext_vector_type(8)))  float  f32x8;

#define BATCH 8192
#define IN_F  2048
#define OUT_F 2048

#define BM 128
#define BN 128
#define BK 64
#define BKP 72   // padded k-stride (bf16 elems) = 144 B = 16*9 -> aligned + conflict-free
#define NTHREADS 512

union FragBits { uint4 q[2]; bf16x16 v; };

// Load one 16x32 bf16 WMMA operand fragment from an LDS tile laid out [row][k]
// with row stride BKP. Per ISA: lane L holds row/col (L&15), k-chunks of 8 at
// (L>>4)*8 and 16+(L>>4)*8 -> two 16-byte ds loads.
__device__ __forceinline__ bf16x16 ldsFrag(const unsigned short* tile, int row, int k0) {
    FragBits f;
    const unsigned short* p = tile + row * BKP + k0;
    f.q[0] = *reinterpret_cast<const uint4*>(p);
    f.q[1] = *reinterpret_cast<const uint4*>(p + 16);
    return f.v;
}

__device__ __forceinline__ uint2 packBf16x4(float a, float b, float c, float d) {
    union { bf16_t h[4]; uint2 u; } pk;
    pk.h[0] = (bf16_t)a; pk.h[1] = (bf16_t)b;
    pk.h[2] = (bf16_t)c; pk.h[3] = (bf16_t)d;
    return pk.u;
}

// Stage a 128x64 fp32 tile PAIR (re, im) from global into three LDS tiles
// (re, im, re+im) as bf16. One global read per element, three 8B LDS stores.
// The sum is formed in fp32 before the single bf16 rounding.
__device__ __forceinline__ void stagePair(const float* __restrict__ gRe,
                                          const float* __restrict__ gIm,
                                          int row0, int kt,
                                          unsigned short* ldsRe,
                                          unsigned short* ldsIm,
                                          unsigned short* ldsS, int t) {
#pragma unroll
    for (int i = 0; i < 4; ++i) {
        const int lin = i * NTHREADS + t;      // 0..2047
        const int r   = lin >> 4;              // 0..127
        const int c   = (lin & 15) << 2;       // 0,4,...,60 (float col)
        const size_t goff = (size_t)(row0 + r) * IN_F + kt + c;
        const float4 a = *reinterpret_cast<const float4*>(gRe + goff);
        const float4 b = *reinterpret_cast<const float4*>(gIm + goff);
        const int loff = r * BKP + c;
        *reinterpret_cast<uint2*>(ldsRe + loff) = packBf16x4(a.x, a.y, a.z, a.w);
        *reinterpret_cast<uint2*>(ldsIm + loff) = packBf16x4(b.x, b.y, b.z, b.w);
        *reinterpret_cast<uint2*>(ldsS  + loff) =
            packBf16x4(a.x + b.x, a.y + b.y, a.z + b.z, a.w + b.w);
    }
}

__global__ __launch_bounds__(NTHREADS)
void cplx_fc_wmma_kernel(const float* __restrict__ xre, const float* __restrict__ xim,
                         const float* __restrict__ wre, const float* __restrict__ wim,
                         float* __restrict__ out) {
    __shared__ __align__(16) unsigned short AsRe[BM * BKP];
    __shared__ __align__(16) unsigned short AsIm[BM * BKP];
    __shared__ __align__(16) unsigned short AsS [BM * BKP];
    __shared__ __align__(16) unsigned short BsRe[BN * BKP];
    __shared__ __align__(16) unsigned short BsIm[BN * BKP];
    __shared__ __align__(16) unsigned short BsS [BN * BKP];

    const int t    = threadIdx.x;
    const int lane = t & 31;
    const int wave = t >> 5;            // 0..15
    const int wm   = wave & 3;          // 4 waves along M, 32 rows each
    const int wn   = wave >> 2;         // 4 waves along N, 32 cols each
    const int lr   = lane & 15;
    const int kb   = (lane >> 4) << 3;  // per-lane k-chunk base: 0 or 8

    const int row0 = blockIdx.y * BM;   // batch rows
    const int col0 = blockIdx.x * BN;   // output features

    f32x8 acc1[2][2];   // x_re * w_re
    f32x8 acc2[2][2];   // x_im * w_im
    f32x8 acc3[2][2];   // (x_re+x_im) * (w_re+w_im)
#pragma unroll
    for (int mi = 0; mi < 2; ++mi)
#pragma unroll
        for (int ni = 0; ni < 2; ++ni) {
            acc1[mi][ni] = {}; acc2[mi][ni] = {}; acc3[mi][ni] = {};
        }

    for (int kt = 0; kt < IN_F; kt += BK) {
        stagePair(xre, xim, row0, kt, AsRe, AsIm, AsS, t);
        stagePair(wre, wim, col0, kt, BsRe, BsIm, BsS, t);
        __syncthreads();

        // Prefetch next K-stage (global_prefetch_b8): 512 threads cover each
        // 128x64 fp32 tile as 64B chunks.
        if (kt + BK < IN_F) {
            const int pr = t >> 2;
            const int pc = (t & 3) * 16;
            const int kn = kt + BK;
            __builtin_prefetch(xre + (size_t)(row0 + pr) * IN_F + kn + pc, 0, 3);
            __builtin_prefetch(xim + (size_t)(row0 + pr) * IN_F + kn + pc, 0, 3);
            __builtin_prefetch(wre + (size_t)(col0 + pr) * IN_F + kn + pc, 0, 3);
            __builtin_prefetch(wim + (size_t)(col0 + pr) * IN_F + kn + pc, 0, 3);
        }

#pragma unroll
        for (int kk = 0; kk < BK; kk += 32) {
            bf16x16 aRe[2], aIm[2], aS[2];
#pragma unroll
            for (int mi = 0; mi < 2; ++mi) {
                const int row = wm * 32 + mi * 16 + lr;
                aRe[mi] = ldsFrag(AsRe, row, kk + kb);
                aIm[mi] = ldsFrag(AsIm, row, kk + kb);
                aS [mi] = ldsFrag(AsS,  row, kk + kb);
            }
#pragma unroll
            for (int ni = 0; ni < 2; ++ni) {
                const int col = wn * 32 + ni * 16 + lr;
                const bf16x16 bRe = ldsFrag(BsRe, col, kk + kb);
                const bf16x16 bIm = ldsFrag(BsIm, col, kk + kb);
                const bf16x16 bS  = ldsFrag(BsS,  col, kk + kb);
#pragma unroll
                for (int mi = 0; mi < 2; ++mi) {
                    acc1[mi][ni] = __builtin_amdgcn_wmma_f32_16x16x32_bf16(
                        false, aRe[mi], false, bRe, (short)0, acc1[mi][ni], false, false);
                    acc2[mi][ni] = __builtin_amdgcn_wmma_f32_16x16x32_bf16(
                        false, aIm[mi], false, bIm, (short)0, acc2[mi][ni], false, false);
                    acc3[mi][ni] = __builtin_amdgcn_wmma_f32_16x16x32_bf16(
                        false, aS[mi],  false, bS,  (short)0, acc3[mi][ni], false, false);
                }
            }
        }
        __syncthreads();
    }

    // Karatsuba recombination + store.
    // C/D layout per ISA: VGPR j holds row (j + 8*(lane>=16)), col lane%16.
    float* __restrict__ outRe = out;
    float* __restrict__ outIm = out + (size_t)BATCH * OUT_F;
    const int rsub = (lane >> 4) << 3;
#pragma unroll
    for (int mi = 0; mi < 2; ++mi) {
#pragma unroll
        for (int ni = 0; ni < 2; ++ni) {
            const f32x8 re = acc1[mi][ni] - acc2[mi][ni];
            const f32x8 im = acc3[mi][ni] - acc1[mi][ni] - acc2[mi][ni];
            const int rbase = row0 + wm * 32 + mi * 16 + rsub;
            const int c     = col0 + wn * 32 + ni * 16 + lr;
#pragma unroll
            for (int j = 0; j < 8; ++j) {
                const size_t off = (size_t)(rbase + j) * OUT_F + c;
                outRe[off] = re[j];
                outIm[off] = im[j];
            }
        }
    }
}

extern "C" void kernel_launch(void* const* d_in, const int* in_sizes, int n_in,
                              void* d_out, int out_size, void* d_ws, size_t ws_size,
                              hipStream_t stream) {
    const float* xre = (const float*)d_in[0];
    const float* xim = (const float*)d_in[1];
    const float* wre = (const float*)d_in[2];
    const float* wim = (const float*)d_in[3];
    float* out = (float*)d_out;

    dim3 grid(OUT_F / BN, BATCH / BM);   // (16, 64)
    dim3 block(NTHREADS);
    cplx_fc_wmma_kernel<<<grid, block, 0, stream>>>(xre, xim, wre, wim, out);
}
